// SelfAttention_16466904613137
// MI455X (gfx1250) — matrix operand order repaired and checked by reading, not compiled, not run on hardware
//
#include <hip/hip_runtime.h>
#include <hip/hip_bf16.h>

#ifndef __has_builtin
#define __has_builtin(x) 0
#endif

#if __has_builtin(__builtin_amdgcn_global_load_async_to_lds_b32) && \
    __has_builtin(__builtin_amdgcn_s_wait_asynccnt)
#define USE_ASYNC_LDS 1
#else
#define USE_ASYNC_LDS 0
#endif

// ---------- WMMA types ----------
typedef __attribute__((ext_vector_type(16))) __bf16 v16bf;
typedef __attribute__((ext_vector_type(8)))  float  v8f;

#if USE_ASYNC_LDS
typedef __attribute__((address_space(1))) int* as1_int_p;
typedef __attribute__((address_space(3))) int* as3_int_p;
#endif

union FragU { unsigned u[8]; v16bf v; };

__device__ inline unsigned short f2bf(float x) {
    unsigned u = __builtin_bit_cast(unsigned, x);
    unsigned r = u + 0x7FFFu + ((u >> 16) & 1u);
    return (unsigned short)(r >> 16);
}
__device__ inline float bf2f(unsigned short s) {
    return __builtin_bit_cast(float, (unsigned)s << 16);
}

__device__ inline v8f wmma_bf16(v16bf a, v16bf b, v8f c) {
    return __builtin_amdgcn_wmma_f32_16x16x32_bf16(
        false, a, false, b, (short)0, c, false, false);
}

// A fragment: 16x32 bf16 from LDS, row stride 34 ushorts (17-uint stride: conflict-free).
__device__ inline v16bf load_frag_a(const unsigned short* As, int lane) {
    int m = lane & 15;
    int half = lane >> 4;
    const unsigned short* row = As + m * 34;
    FragU f;
#pragma unroll
    for (int j = 0; j < 8; ++j) {
        int k = ((j >> 2) << 4) + (half << 3) + ((j & 3) << 1);
        f.u[j] = *(const unsigned*)(row + k);
    }
    return f.v;
}

// B fragment: 32x16, staged n-major (Bs[n][k], stride 34).
__device__ inline v16bf load_frag_b(const unsigned short* Bs, int lane) {
    int n = lane & 15;
    int half = lane >> 4;
    const unsigned short* row = Bs + n * 34 + (half << 3);
    FragU f;
#pragma unroll
    for (int j = 0; j < 8; ++j) f.u[j] = *(const unsigned*)(row + ((j >> 2) << 4) + ((j & 3) << 1));
    return f.v;
}

// Stage a 128-row x 32-k bf16 tile (contiguous along k in memory) into LDS stride-34.
__device__ inline void stage_tile(unsigned short* lds, const unsigned short* src,
                                  int ld, int k0, int tid) {
#if USE_ASYNC_LDS
    // gfx1250 async DMA: memory -> LDS, tracked by ASYNCcnt (8 ops/thread)
#pragma unroll
    for (int i = 0; i < 8; ++i) {
        int idx = tid + i * 256;
        int r = idx >> 4, cu = idx & 15;
        const unsigned short* g = src + (size_t)r * ld + k0 + (cu << 1);
        unsigned short* l = lds + r * 34 + (cu << 1);
        __builtin_amdgcn_global_load_async_to_lds_b32(
            (as1_int_p)(void*)g, (as3_int_p)(void*)l, 0, 0);
    }
#else
    // fallback: batch ALL global loads (b128) before ANY LDS store
    const uint4* g4 = (const uint4*)(src + k0);
    int ld4 = ld >> 3;
    uint4 t0, t1;
    {
        int idx = tid;
        t0 = g4[(size_t)(idx >> 2) * ld4 + (idx & 3)];
        idx = tid + 256;
        t1 = g4[(size_t)(idx >> 2) * ld4 + (idx & 3)];
    }
    {
        int idx = tid;
        unsigned* d = (unsigned*)(lds + (idx >> 2) * 34 + ((idx & 3) << 3));
        d[0] = t0.x; d[1] = t0.y; d[2] = t0.z; d[3] = t0.w;
        idx = tid + 256;
        d = (unsigned*)(lds + (idx >> 2) * 34 + ((idx & 3) << 3));
        d[0] = t1.x; d[1] = t1.y; d[2] = t1.z; d[3] = t1.w;
    }
#endif
}

// ---------- cast kernels ----------
__global__ __launch_bounds__(256)
void cast_f32_bf16(const float* __restrict__ in, unsigned short* __restrict__ out,
                   long long n) {
    long long i = (long long)blockIdx.x * blockDim.x + threadIdx.x;
    long long stride = (long long)gridDim.x * blockDim.x;
    for (; i < n; i += stride) out[i] = f2bf(in[i]);
}

// in [rows][cols] f32 -> out [cols][rows] bf16 (LDS-tiled 32x32)
__global__ __launch_bounds__(256)
void cast_transpose_f32_bf16(const float* __restrict__ in, unsigned short* __restrict__ out,
                             int rows, int cols) {
    __shared__ float tile[32][33];
    int bx = blockIdx.x * 32;   // col base
    int by = blockIdx.y * 32;   // row base
    int tx = threadIdx.x & 31, ty = threadIdx.x >> 5;
#pragma unroll
    for (int i = 0; i < 32; i += 8)
        tile[ty + i][tx] = in[(size_t)(by + ty + i) * cols + bx + tx];
    __syncthreads();
#pragma unroll
    for (int i = 0; i < 32; i += 8)
        out[(size_t)(bx + ty + i) * rows + by + tx] = f2bf(tile[tx][ty + i]);
}

// ---------- WMMA GEMM: C = A @ Bmat^T, 128x128 tile, 8 waves, double-buffered LDS ----
// A: [M][Kd] bf16, lda.  Bmat: [N][Kd] bf16, ldb (logical B = Bmat^T, contiguous along k).
// EPI 0: bf16 out (+bias if BIAS); if STORET, write C^T (ldc = leading dim of C^T)
// EPI 1: scale + inverted-causal mask -> bf16
// EPI 2: f32 out + bias
template<int EPI, bool BIAS, bool STORET>
__global__ __launch_bounds__(256)
void gemm_wmma(const unsigned short* __restrict__ A,
               const unsigned short* __restrict__ B,
               void* __restrict__ Cv,
               const float* __restrict__ bias,
               int Kd, int lda, int ldb, int ldc,
               unsigned long long strideA, unsigned long long strideB,
               unsigned long long strideC, float scale) {
    __shared__ __align__(16) unsigned short As[2][128 * 34];
    __shared__ __align__(16) unsigned short Bs[2][128 * 34];

    int tid  = threadIdx.x;
    int lane = tid & 31;
    int wave = tid >> 5;
    int wm = (wave >> 1) * 32;   // 0,32,64,96
    int wn = (wave & 1) * 64;    // 0,64

    unsigned long long zb = blockIdx.z;
    const unsigned short* Asrc = A + zb * strideA + (size_t)blockIdx.y * 128 * lda;
    const unsigned short* Bsrc = B + zb * strideB + (size_t)blockIdx.x * 128 * ldb;
    int row0 = blockIdx.y * 128;
    int col0 = blockIdx.x * 128;

    v8f acc[2][4] = {};

    int ksteps = Kd >> 5;
    stage_tile(As[0], Asrc, lda, 0, tid);
    stage_tile(Bs[0], Bsrc, ldb, 0, tid);
    int buf = 0;
    for (int kt = 0; kt < ksteps; ++kt) {
        if (kt + 1 < ksteps) {
            int kn = (kt + 1) << 5;
            stage_tile(As[buf ^ 1], Asrc, lda, kn, tid);
            stage_tile(Bs[buf ^ 1], Bsrc, ldb, kn, tid);
#if USE_ASYNC_LDS
            __builtin_amdgcn_s_wait_asynccnt(16);  // current buf's 16 ops complete
#endif
        } else {
#if USE_ASYNC_LDS
            __builtin_amdgcn_s_wait_asynccnt(0);
#endif
        }
        if (kt + 2 < ksteps) {
            const char* pa = (const char*)Asrc +
                (((size_t)(tid >> 1) * lda + ((size_t)(kt + 2) << 5) + ((tid & 1) << 4)) << 1);
            __builtin_prefetch(pa, 0, 0);          // gfx1250 global_prefetch_b8
        }
        __syncthreads();                            // buf tiles visible to all waves
        const unsigned short* as = As[buf];
        const unsigned short* bs = Bs[buf];
        v16bf a0 = load_frag_a(as + wm * 34, lane);
        v16bf a1 = load_frag_a(as + (wm + 16) * 34, lane);
        v16bf b0 = load_frag_b(bs + (wn +  0) * 34, lane);
        v16bf b1 = load_frag_b(bs + (wn + 16) * 34, lane);
        v16bf b2 = load_frag_b(bs + (wn + 32) * 34, lane);
        v16bf b3 = load_frag_b(bs + (wn + 48) * 34, lane);
        acc[0][0] = wmma_bf16(a0, b0, acc[0][0]);
        acc[0][1] = wmma_bf16(a0, b1, acc[0][1]);
        acc[0][2] = wmma_bf16(a0, b2, acc[0][2]);
        acc[0][3] = wmma_bf16(a0, b3, acc[0][3]);
        acc[1][0] = wmma_bf16(a1, b0, acc[1][0]);
        acc[1][1] = wmma_bf16(a1, b1, acc[1][1]);
        acc[1][2] = wmma_bf16(a1, b2, acc[1][2]);
        acc[1][3] = wmma_bf16(a1, b3, acc[1][3]);
        __syncthreads();                            // all reads done before restage
        buf ^= 1;
    }

    // ---- epilogue: C layout VGPR j -> M=j / j+8 per lane-half, N=lane%16 ----
    int half = lane >> 4, nl = lane & 15;
    float bv[4] = {0.f, 0.f, 0.f, 0.f};
    if (BIAS) {
#pragma unroll
        for (int ni = 0; ni < 4; ++ni) bv[ni] = bias[col0 + wn + ni * 16 + nl];
    }
#pragma unroll
    for (int mi = 0; mi < 2; ++mi) {
        int rbase = row0 + wm + mi * 16 + half * 8;
#pragma unroll
        for (int ni = 0; ni < 4; ++ni) {
            int c = col0 + wn + ni * 16 + nl;
            if (STORET) {
                // C^T[c][rbase..rbase+7]: 8 contiguous bf16 -> single b128 store
                unsigned short* pt = (unsigned short*)Cv + zb * strideC +
                                     (size_t)c * ldc + rbase;
                uint4 pk;
                unsigned w[4];
#pragma unroll
                for (int jj = 0; jj < 4; ++jj) {
                    float v0 = acc[mi][ni][2 * jj]     + bv[ni];
                    float v1 = acc[mi][ni][2 * jj + 1] + bv[ni];
                    w[jj] = (unsigned)f2bf(v0) | ((unsigned)f2bf(v1) << 16);
                }
                pk.x = w[0]; pk.y = w[1]; pk.z = w[2]; pk.w = w[3];
                *(uint4*)pt = pk;
            } else if (EPI == 2) {
                float* p = (float*)Cv + zb * strideC + (size_t)rbase * ldc + c;
#pragma unroll
                for (int j = 0; j < 8; ++j) { p[0] = acc[mi][ni][j] + bv[ni]; p += ldc; }
            } else {
                unsigned short* p = (unsigned short*)Cv + zb * strideC +
                                    (size_t)rbase * ldc + c;
#pragma unroll
                for (int j = 0; j < 8; ++j) {
                    float v = acc[mi][ni][j];
                    if (EPI == 1) { v *= scale; if (c <= rbase + j + 1) v = -1e9f; }
                    else if (BIAS) v += bv[ni];
                    p[0] = f2bf(v);
                    p += ldc;
                }
            }
        }
    }
}

// ---------- row softmax (2048-wide rows, in place, bf16) ----------
__global__ __launch_bounds__(256)
void softmax_rows2048(unsigned short* __restrict__ S) {
    __shared__ float red[256];
    size_t base = (size_t)blockIdx.x * 2048;
    int tid = threadIdx.x;
    float v[8];
#pragma unroll
    for (int i = 0; i < 8; ++i) v[i] = bf2f(S[base + tid + i * 256]);
    float m = v[0];
#pragma unroll
    for (int i = 1; i < 8; ++i) m = fmaxf(m, v[i]);
    red[tid] = m; __syncthreads();
    for (int s = 128; s > 0; s >>= 1) {
        if (tid < s) red[tid] = fmaxf(red[tid], red[tid + s]);
        __syncthreads();
    }
    m = red[0]; __syncthreads();
    float sum = 0.f;
#pragma unroll
    for (int i = 0; i < 8; ++i) { v[i] = __expf(v[i] - m); sum += v[i]; }
    red[tid] = sum; __syncthreads();
    for (int s = 128; s > 0; s >>= 1) {
        if (tid < s) red[tid] += red[tid + s];
        __syncthreads();
    }
    float inv = 1.f / red[0];
#pragma unroll
    for (int i = 0; i < 8; ++i) S[base + tid + i * 256] = f2bf(v[i] * inv);
}

// ---------- driver ----------
extern "C" void kernel_launch(void* const* d_in, const int* in_sizes, int n_in,
                              void* d_out, int out_size, void* d_ws, size_t ws_size,
                              hipStream_t stream) {
    const float* x  = (const float*)d_in[0];
    const float* wq = (const float*)d_in[1];
    const float* bq = (const float*)d_in[2];
    const float* wk = (const float*)d_in[3];
    const float* bk = (const float*)d_in[4];
    const float* wv = (const float*)d_in[5];
    const float* bv = (const float*)d_in[6];
    const float* wo = (const float*)d_in[7];
    const float* bo = (const float*)d_in[8];
    float* out = (float*)d_out;

    const int B = 4, S = 2048, D = 1024, K = 1024;
    const size_t MS = (size_t)B * S;        // 8192 rows

    char* ws = (char*)d_ws;
    size_t off = 0;
    auto alloc = [&](size_t bytes) -> void* {
        void* p = ws + off;
        off += (bytes + 255) & ~(size_t)255;
        return p;
    };
    unsigned short* Xb  = (unsigned short*)alloc(MS * D * 2);
    unsigned short* WqT = (unsigned short*)alloc((size_t)D * K * 2);  // [K][D]
    unsigned short* WkT = (unsigned short*)alloc((size_t)D * K * 2);  // [K][D]
    unsigned short* WvT = (unsigned short*)alloc((size_t)D * K * 2);  // [K][D]
    unsigned short* WoT = (unsigned short*)alloc((size_t)K * D * 2);  // [D][K]
    unsigned short* Qb  = (unsigned short*)alloc(MS * K * 2);         // [MS][K]
    unsigned short* Kb  = (unsigned short*)alloc(MS * K * 2);         // [MS][K]
    unsigned short* Vt  = (unsigned short*)alloc(MS * K * 2);         // [K][MS] (V^T)
    unsigned short* Sb  = (unsigned short*)alloc((size_t)B * S * S * 2);
    unsigned short* Ob  = (unsigned short*)alloc(MS * K * 2);         // [MS][K]

    // 1) casts: x plain; weights cast+transpose -> [N][Kd] for contiguous-k staging
    cast_f32_bf16<<<4096, 256, 0, stream>>>(x, Xb, (long long)MS * D);
    dim3 gt(K / 32, D / 32);
    cast_transpose_f32_bf16<<<gt, 256, 0, stream>>>(wq, WqT, D, K);
    cast_transpose_f32_bf16<<<gt, 256, 0, stream>>>(wk, WkT, D, K);
    cast_transpose_f32_bf16<<<gt, 256, 0, stream>>>(wv, WvT, D, K);
    cast_transpose_f32_bf16<<<gt, 256, 0, stream>>>(wo, WoT, K, D);

    dim3 blk(256);
    // 2) projections [8192 x 1024] = Xb @ W^T^T  (B given as Wt[N][D])
    dim3 gproj(K / 128, (unsigned)(MS / 128), 1);
    gemm_wmma<0, true, false><<<gproj, blk, 0, stream>>>(Xb, WqT, Qb, bq,
        D, D, D, K, 0ull, 0ull, 0ull, 1.0f);
    gemm_wmma<0, true, false><<<gproj, blk, 0, stream>>>(Xb, WkT, Kb, bk,
        D, D, D, K, 0ull, 0ull, 0ull, 1.0f);
    // V projection writes V^T directly: Vt[K][MS], ldc = MS
    gemm_wmma<0, true, true><<<gproj, blk, 0, stream>>>(Xb, WvT, Vt, bv,
        D, D, D, (int)MS, 0ull, 0ull, 0ull, 1.0f);

    // 3) scores = (Q @ K^T)/32 + inverted-causal mask, per batch
    dim3 gsc(S / 128, S / 128, B);
    gemm_wmma<1, false, false><<<gsc, blk, 0, stream>>>(Qb, Kb, Sb, nullptr,
        K, K, K, S,
        (unsigned long long)S * K, (unsigned long long)S * K,
        (unsigned long long)S * S, 1.0f / 32.0f);

    // 4) softmax over each of the 8192 score rows (in place)
    softmax_rows2048<<<(unsigned)MS, 256, 0, stream>>>(Sb);

    // 5) attn @ V per batch: A = P[b] [S x S], Bmat = V^T columns for batch b
    //    (Vt[K][MS]; batch b's submatrix has row stride MS, element offset b*S)
    dim3 gav(K / 128, S / 128, B);
    gemm_wmma<0, false, false><<<gav, blk, 0, stream>>>(Sb, Vt, Ob, nullptr,
        S, S, (int)MS, K,
        (unsigned long long)S * S, (unsigned long long)S,
        (unsigned long long)S * K, 1.0f);

    // 6) out = Ob @ Wo + bo -> f32 d_out [8192 x 1024]
    dim3 gout(D / 128, (unsigned)(MS / 128), 1);
    gemm_wmma<2, true, false><<<gout, blk, 0, stream>>>(Ob, WoT, out, bo,
        K, K, K, D, 0ull, 0ull, 0ull, 1.0f);
}
